// CompositeGNNnodeBased_8306466750475
// MI455X (gfx1250) — matrix operand order repaired and checked by reading, not compiled, not run on hardware
//
#include <hip/hip_runtime.h>

// ---------------------------------------------------------------- constants
#define NN   100000      // nodes
#define EE   1600000     // edges
#define DD   64          // node dim = state dim
#define HH   128         // hidden
#define K1P  352         // 328 padded to 11*32
#define NHID 256         // hidden for both types fused
#define OUTD 8

typedef __attribute__((ext_vector_type(16))) __bf16 v16bf;
typedef __attribute__((ext_vector_type(8)))  __bf16 v8bf;
typedef __attribute__((ext_vector_type(8)))  float  v8f;

static __device__ __forceinline__ __bf16 f2bf(float f) {
  unsigned u = __builtin_bit_cast(unsigned, f);
  unsigned r = (u + 0x7FFFu + ((u >> 16) & 1u)) >> 16;
  unsigned short s = (unsigned short)r;
  return __builtin_bit_cast(__bf16, s);
}

// ---------------------------------------------------------------- init
__global__ void gnn_init_kernel(const float* __restrict__ sinit,
                                float* __restrict__ bufA, float* __restrict__ bufB,
                                int* __restrict__ flags, int total) {
  int i = blockIdx.x * blockDim.x + threadIdx.x;
  if (i == 0) { flags[0] = 0; flags[1] = 0; }   // stop, anyflag
  if (i < total) { bufA[i] = sinit[i]; bufB[i] = 1.0f; }
}

// ------------------------------------------------- weight transpose -> bf16
__global__ void gnn_convw_kernel(const float* __restrict__ Ws0_1, const float* __restrict__ Ws1_1,
                                 const float* __restrict__ Ws0_2, const float* __restrict__ Ws1_2,
                                 const float* __restrict__ Wo_1,  const float* __restrict__ Wo_2,
                                 __bf16* __restrict__ W1t, __bf16* __restrict__ W2t,
                                 __bf16* __restrict__ Wo1t, __bf16* __restrict__ Wo2t) {
  int i = blockIdx.x * blockDim.x + threadIdx.x;
  const int R1 = NHID * K1P;            // 90112  : W1t[n][k], n<128 type0, else type1
  const int R2 = R1 + 2 * DD * HH;      // +16384 : W2t[t][n][k]
  const int R3 = R2 + HH * DD;          // +8192  : Wo1t[n][k]
  const int R4 = R3 + 16 * HH;          // +2048  : Wo2t[n][k] (rows 8..15 zero)
  if (i < R1) {
    int n = i / K1P, k = i % K1P;
    float v = 0.f;
    if (k < 328) v = (n < HH) ? Ws0_1[k * HH + n] : Ws1_1[k * HH + (n - HH)];
    W1t[i] = f2bf(v);
  } else if (i < R2) {
    int j = i - R1; int t = j / (DD * HH); int n = (j / HH) % DD; int k = j % HH;
    const float* W = t ? Ws1_2 : Ws0_2;
    W2t[j] = f2bf(W[k * DD + n]);
  } else if (i < R3) {
    int j = i - R2; int n = j / DD; int k = j % DD;
    Wo1t[j] = f2bf(Wo_1[k * HH + n]);
  } else if (i < R4) {
    int j = i - R3; int n = j / HH; int k = j % HH;
    Wo2t[j] = f2bf(n < OUTD ? Wo_2[k * OUTD + n] : 0.f);
  }
}

// ------------------------------------------------- one-time edge aggregates
__global__ void gnn_scatter_comp_kernel(const float* __restrict__ nodes,
                                        const float* __restrict__ c0, const float* __restrict__ c1,
                                        const int* __restrict__ src, const int* __restrict__ dst,
                                        float* __restrict__ a0, float* __restrict__ a1) {
  int i = blockIdx.x * blockDim.x + threadIdx.x;
  if (i >= EE * 64) return;
  int e = i >> 6, c = i & 63;
  float v = nodes[src[e] * 64 + c];
  int o = dst[e] * 64 + c;
  atomicAdd(&a0[o], c0[e] * v);
  atomicAdd(&a1[o], c1[e] * v);
}

__global__ void gnn_scatter_arcs_kernel(const float* __restrict__ arcs, const float* __restrict__ an,
                                        const int* __restrict__ dst, float* __restrict__ aa) {
  int i = blockIdx.x * blockDim.x + threadIdx.x;
  if (i >= EE * 8) return;
  int e = i >> 3, c = i & 7;
  atomicAdd(&aa[dst[e] * 8 + c], an[e] * arcs[e * 8 + c]);
}

// ------------------------------------------------- per-iteration scatter
__global__ void gnn_scatter_state_kernel(const float* __restrict__ s, const float* __restrict__ adj,
                                         const int* __restrict__ src, const int* __restrict__ dst,
                                         float* __restrict__ ast, const int* __restrict__ flags) {
  if (flags[0]) return;   // converged: skip
  int i = blockIdx.x * blockDim.x + threadIdx.x;
  if (i >= EE * 64) return;
  int e = i >> 6, c = i & 63;
  atomicAdd(&ast[dst[e] * 64 + c], adj[e] * s[src[e] * 64 + c]);
}

// ------------------------------------------------- convergence cond + gate
__global__ void gnn_cond_kernel(const float* __restrict__ s, const float* __restrict__ so,
                                int* __restrict__ flags) {
  if (flags[0]) return;
  int n = blockIdx.x * blockDim.x + threadIdx.x;
  if (n >= NN) return;
  float d2 = 0.f, n2 = 0.f;
  #pragma unroll 4
  for (int c = 0; c < 64; ++c) {
    float a = s[n * 64 + c], b = so[n * 64 + c];
    float df = a - b;
    d2 += df * df;
    n2 += b * b;
  }
  if (d2 > 1e-4f * n2) flags[1] = 1;   // THRESHOLD^2 = 0.0001
}

__global__ void gnn_gate_kernel(int* __restrict__ flags) {
  if (flags[1] == 0) flags[0] = 1;     // nobody moved -> stop
  flags[1] = 0;
}

// ------------------------------------------------- fused per-type MLP (WMMA)
// wave32 per 16-node tile: inp[16,352]bf16 @ W1t[352,256] -> tanh -> LDS ->
// h[16,256] @ W2t -> type-select -> s_out[16,64]
__global__ __launch_bounds__(32)
void gnn_mlp_kernel(const float* __restrict__ nodes, const float* __restrict__ s_in,
                    const float* __restrict__ aggst, const float* __restrict__ agg0,
                    const float* __restrict__ agg1,  const float* __restrict__ aggarc,
                    const __bf16* __restrict__ W1t,  const float* __restrict__ b1a,
                    const float* __restrict__ b1b,   const __bf16* __restrict__ W2t,
                    const float* __restrict__ b2a,   const float* __restrict__ b2b,
                    const int* __restrict__ type_id, const int* __restrict__ flags,
                    float* __restrict__ s_out) {
  const int tile = blockIdx.x;
  const int lane = threadIdx.x;
  const int row  = lane & 15;
  const int hi   = lane >> 4;
  const int node = tile * 16 + row;

  if (flags[0]) {   // converged: forward state so host-side buffer swap stays valid
    #pragma unroll
    for (int j = 0; j < 8; ++j) {
      int c = hi * 32 + j * 4;
      *(float4*)&s_out[node * 64 + c] = *(const float4*)&s_in[node * 64 + c];
    }
    return;
  }

  __shared__ __bf16 hlds[16 * 272];   // 16 rows x 256 hidden (+16 pad)

  // -------- GEMM1: inp[16,352] x W1t -> acc[16x256] (both types fused)
  v8f acc[16] = {};
  #pragma unroll
  for (int kc = 0; kc < 11; ++kc) {
    const int kb = kc * 32;
    v16bf a;
    #pragma unroll
    for (int half = 0; half < 2; ++half) {
      const int c = kb + hi * 16 + half * 8;   // 8-run start col, always inside one region
      const float* p = nullptr;
      if      (c < 64)  p = nodes  + node * 64 + c;
      else if (c < 128) p = s_in   + node * 64 + (c - 64);
      else if (c < 192) p = aggst  + node * 64 + (c - 128);
      else if (c < 256) p = agg0   + node * 64 + (c - 192);
      else if (c < 320) p = agg1   + node * 64 + (c - 256);
      else if (c < 328) p = aggarc + node * 8  + (c - 320);
      if (p) {
        float4 v0 = *(const float4*)p;
        float4 v1 = *(const float4*)(p + 4);
        a[half * 8 + 0] = f2bf(v0.x); a[half * 8 + 1] = f2bf(v0.y);
        a[half * 8 + 2] = f2bf(v0.z); a[half * 8 + 3] = f2bf(v0.w);
        a[half * 8 + 4] = f2bf(v1.x); a[half * 8 + 5] = f2bf(v1.y);
        a[half * 8 + 6] = f2bf(v1.z); a[half * 8 + 7] = f2bf(v1.w);
      } else {
        #pragma unroll
        for (int j = 0; j < 8; ++j) a[half * 8 + j] = f2bf(0.f);
      }
    }
    #pragma unroll
    for (int nt = 0; nt < 16; ++nt) {
      const __bf16* bp = W1t + (size_t)(nt * 16 + row) * K1P + kb + hi * 16;
      v16bf b = *(const v16bf*)bp;
      acc[nt] = __builtin_amdgcn_wmma_f32_16x16x32_bf16(false, a, false, b,
                                                        (short)0, acc[nt], false, false);
    }
  }

  // -------- bias + tanh, stage hidden as bf16 in LDS
  #pragma unroll
  for (int nt = 0; nt < 16; ++nt) {
    const int n = nt * 16 + row;
    const float bias = (n < HH) ? b1a[n] : b1b[n - HH];
    #pragma unroll
    for (int r = 0; r < 8; ++r) {
      float v = tanhf(acc[nt][r] + bias);
      const int m = r + hi * 8;
      hlds[m * 272 + n] = f2bf(v);
    }
  }
  __syncthreads();

  // -------- GEMM2: h0/h1[16,128] x W2t[t] -> 16x64 per type
  v8f o0[4] = {}, o1[4] = {};
  #pragma unroll
  for (int kc = 0; kc < 4; ++kc) {
    const int kb = kc * 32;
    const int koff = kb + hi * 16;
    v8bf r0 = *(const v8bf*)&hlds[row * 272 + koff];
    v8bf r1 = *(const v8bf*)&hlds[row * 272 + koff + 8];
    v8bf r2 = *(const v8bf*)&hlds[row * 272 + 128 + koff];
    v8bf r3 = *(const v8bf*)&hlds[row * 272 + 128 + koff + 8];
    v16bf a0 = __builtin_shufflevector(r0, r1, 0,1,2,3,4,5,6,7,8,9,10,11,12,13,14,15);
    v16bf a1 = __builtin_shufflevector(r2, r3, 0,1,2,3,4,5,6,7,8,9,10,11,12,13,14,15);
    #pragma unroll
    for (int nt = 0; nt < 4; ++nt) {
      const __bf16* b0p = W2t + (size_t)(nt * 16 + row) * HH + kb + hi * 16;
      const __bf16* b1p = b0p + (size_t)DD * HH;
      v16bf b0 = *(const v16bf*)b0p;
      v16bf b1 = *(const v16bf*)b1p;
      o0[nt] = __builtin_amdgcn_wmma_f32_16x16x32_bf16(false, a0, false, b0,
                                                       (short)0, o0[nt], false, false);
      o1[nt] = __builtin_amdgcn_wmma_f32_16x16x32_bf16(false, a1, false, b1,
                                                       (short)0, o1[nt], false, false);
    }
  }

  // -------- per-node type select + bias + store
  int tl[8];
  #pragma unroll
  for (int r = 0; r < 8; ++r) tl[r] = type_id[tile * 16 + hi * 8 + r];
  #pragma unroll
  for (int nt = 0; nt < 4; ++nt) {
    const int n = nt * 16 + row;
    const float bza = b2a[n], bzb = b2b[n];
    #pragma unroll
    for (int r = 0; r < 8; ++r) {
      const int m = hi * 8 + r;
      const float v = (tl[r] == 0) ? (o0[nt][r] + bza) : (o1[nt][r] + bzb);
      s_out[(tile * 16 + m) * 64 + n] = v;
    }
  }
}

// ------------------------------------------------- output MLP (WMMA)
__global__ __launch_bounds__(32)
void gnn_out_kernel(const float* __restrict__ state, const __bf16* __restrict__ Wo1t,
                    const float* __restrict__ bo1, const __bf16* __restrict__ Wo2t,
                    const float* __restrict__ bo2, const unsigned char* __restrict__ smask,
                    const unsigned char* __restrict__ omask, float* __restrict__ out) {
  const int tile = blockIdx.x;
  const int lane = threadIdx.x;
  const int row  = lane & 15;
  const int hi   = lane >> 4;
  const int node = tile * 16 + row;

  __shared__ __bf16 h[16 * 136];   // 16 x 128 (+8 pad)

  v8f acc[8] = {};
  #pragma unroll
  for (int kc = 0; kc < 2; ++kc) {
    const int kb = kc * 32;
    v16bf a;
    #pragma unroll
    for (int half = 0; half < 2; ++half) {
      const int c = kb + hi * 16 + half * 8;
      const float* p = state + node * 64 + c;
      float4 v0 = *(const float4*)p;
      float4 v1 = *(const float4*)(p + 4);
      a[half * 8 + 0] = f2bf(v0.x); a[half * 8 + 1] = f2bf(v0.y);
      a[half * 8 + 2] = f2bf(v0.z); a[half * 8 + 3] = f2bf(v0.w);
      a[half * 8 + 4] = f2bf(v1.x); a[half * 8 + 5] = f2bf(v1.y);
      a[half * 8 + 6] = f2bf(v1.z); a[half * 8 + 7] = f2bf(v1.w);
    }
    #pragma unroll
    for (int nt = 0; nt < 8; ++nt) {
      const __bf16* bp = Wo1t + (size_t)(nt * 16 + row) * DD + kb + hi * 16;
      v16bf b = *(const v16bf*)bp;
      acc[nt] = __builtin_amdgcn_wmma_f32_16x16x32_bf16(false, a, false, b,
                                                        (short)0, acc[nt], false, false);
    }
  }

  #pragma unroll
  for (int nt = 0; nt < 8; ++nt) {
    const int n = nt * 16 + row;
    const float bias = bo1[n];
    #pragma unroll
    for (int r = 0; r < 8; ++r) {
      float v = tanhf(acc[nt][r] + bias);
      h[(r + hi * 8) * 136 + n] = f2bf(v);
    }
  }
  __syncthreads();

  v8f o = {};
  #pragma unroll
  for (int kc = 0; kc < 4; ++kc) {
    const int kb = kc * 32;
    const int koff = kb + hi * 16;
    v8bf r0 = *(const v8bf*)&h[row * 136 + koff];
    v8bf r1 = *(const v8bf*)&h[row * 136 + koff + 8];
    v16bf a = __builtin_shufflevector(r0, r1, 0,1,2,3,4,5,6,7,8,9,10,11,12,13,14,15);
    const __bf16* bp = Wo2t + (size_t)row * HH + kb + hi * 16;
    v16bf b = *(const v16bf*)bp;
    o = __builtin_amdgcn_wmma_f32_16x16x32_bf16(false, a, false, b, (short)0, o, false, false);
  }

  if (row < OUTD) {
    const float bias = bo2[row];
    #pragma unroll
    for (int r = 0; r < 8; ++r) {
      const int nd = tile * 16 + hi * 8 + r;
      const float m = (smask[nd] && omask[nd]) ? 1.f : 0.f;
      out[nd * OUTD + row] = (o[r] + bias) * m;
    }
  }
}

// ---------------------------------------------------------------- launch
extern "C" void kernel_launch(void* const* d_in, const int* in_sizes, int n_in,
                              void* d_out, int out_size, void* d_ws, size_t ws_size,
                              hipStream_t stream) {
  (void)in_sizes; (void)n_in; (void)out_size; (void)ws_size;

  const float* nodes      = (const float*)d_in[0];
  const float* arcs_feat  = (const float*)d_in[1];
  const float* state_init = (const float*)d_in[2];
  const float* adj_vals   = (const float*)d_in[3];
  const float* comp0      = (const float*)d_in[4];
  const float* comp1      = (const float*)d_in[5];
  const float* an_vals    = (const float*)d_in[6];
  const float* Ws0_1 = (const float*)d_in[7];
  const float* bs0_1 = (const float*)d_in[8];
  const float* Ws0_2 = (const float*)d_in[9];
  const float* bs0_2 = (const float*)d_in[10];
  const float* Ws1_1 = (const float*)d_in[11];
  const float* bs1_1 = (const float*)d_in[12];
  const float* Ws1_2 = (const float*)d_in[13];
  const float* bs1_2 = (const float*)d_in[14];
  const float* Wo_1  = (const float*)d_in[15];
  const float* bo_1  = (const float*)d_in[16];
  const float* Wo_2  = (const float*)d_in[17];
  const float* bo_2  = (const float*)d_in[18];
  const int* edge_src = (const int*)d_in[19];
  const int* edge_dst = (const int*)d_in[20];
  const int* type_id  = (const int*)d_in[21];
  const unsigned char* set_mask = (const unsigned char*)d_in[22];
  const unsigned char* out_mask = (const unsigned char*)d_in[23];

  // workspace carve (all sizes multiples of 256B)
  char* w = (char*)d_ws;
  size_t off = 0;
  auto take = [&](size_t b) { void* p = w + off; off += (b + 255) & ~(size_t)255; return p; };
  const size_t N64 = (size_t)NN * 64;
  float*  bufA   = (float*)take(N64 * 4);
  float*  bufB   = (float*)take(N64 * 4);
  float*  aggst  = (float*)take(N64 * 4);
  float*  agg0   = (float*)take(N64 * 4);
  float*  agg1   = (float*)take(N64 * 4);
  float*  aggarc = (float*)take((size_t)NN * 8 * 4);
  __bf16* W1t    = (__bf16*)take((size_t)NHID * K1P * 2);
  __bf16* W2t    = (__bf16*)take((size_t)2 * DD * HH * 2);
  __bf16* Wo1t   = (__bf16*)take((size_t)HH * DD * 2);
  __bf16* Wo2t   = (__bf16*)take((size_t)16 * HH * 2);
  int*    flags  = (int*)take(256);

  hipMemsetAsync(agg0,   0, N64 * 4, stream);
  hipMemsetAsync(agg1,   0, N64 * 4, stream);
  hipMemsetAsync(aggarc, 0, (size_t)NN * 8 * 4, stream);

  gnn_init_kernel<<<(int)((N64 + 255) / 256), 256, 0, stream>>>(state_init, bufA, bufB, flags, (int)N64);
  gnn_convw_kernel<<<(116736 + 255) / 256, 256, 0, stream>>>(Ws0_1, Ws1_1, Ws0_2, Ws1_2, Wo_1, Wo_2,
                                                             W1t, W2t, Wo1t, Wo2t);
  gnn_scatter_comp_kernel<<<EE * 64 / 256, 256, 0, stream>>>(nodes, comp0, comp1, edge_src, edge_dst,
                                                             agg0, agg1);
  gnn_scatter_arcs_kernel<<<EE * 8 / 256, 256, 0, stream>>>(arcs_feat, an_vals, edge_dst, aggarc);

  for (int k = 0; k < 10; ++k) {
    const float* sin  = (k % 2 == 0) ? bufA : bufB;
    float*       sout = (k % 2 == 0) ? bufB : bufA;
    gnn_cond_kernel<<<(NN + 255) / 256, 256, 0, stream>>>(sin, sout /* previous state */, flags);
    gnn_gate_kernel<<<1, 1, 0, stream>>>(flags);
    hipMemsetAsync(aggst, 0, N64 * 4, stream);
    gnn_scatter_state_kernel<<<EE * 64 / 256, 256, 0, stream>>>(sin, adj_vals, edge_src, edge_dst,
                                                                aggst, flags);
    gnn_mlp_kernel<<<NN / 16, 32, 0, stream>>>(nodes, sin, aggst, agg0, agg1, aggarc,
                                               W1t, bs0_1, bs1_1, W2t, bs0_2, bs1_2,
                                               type_id, flags, sout);
  }

  // after 10 iterations (k=9 wrote bufA), final state is in bufA
  gnn_out_kernel<<<NN / 16, 32, 0, stream>>>(bufA, Wo1t, bo_1, Wo2t, bo_2,
                                             set_mask, out_mask, (float*)d_out);
}
